// LSTM_49014166782465
// MI455X (gfx1250) — matrix operand order repaired and checked by reading, not compiled, not run on hardware
//
#include <hip/hip_runtime.h>
#include <hip/hip_bf16.h>

#define L_SEQ 4096
#define HID   1024
#define G4H   4096
#define NTAG  50
#define NWG   64          // persistent workgroups for recurrence

typedef __attribute__((ext_vector_type(16))) __bf16 v16bf;
typedef __attribute__((ext_vector_type(8)))  float  v8f;

__device__ __forceinline__ unsigned short f2bf(float f) {
    unsigned int u = __float_as_uint(f);
    unsigned int r = u + 0x7FFFu + ((u >> 16) & 1u);   // round-to-nearest-even
    return (unsigned short)(r >> 16);
}

// ---------------------------------------------------------------------------
// Kernel 1: zero sync state, convert W_ih -> bf16, gather+convert emb[tokens]
// ---------------------------------------------------------------------------
__global__ void __launch_bounds__(256)
prep_kernel(const int* __restrict__ tokens, const float* __restrict__ emb,
            const float* __restrict__ W_ih,
            unsigned short* __restrict__ xbf, unsigned short* __restrict__ wbf,
            float* __restrict__ hbuf, unsigned int* __restrict__ counter)
{
    const size_t i      = (size_t)blockIdx.x * blockDim.x + threadIdx.x;
    const size_t stride = (size_t)gridDim.x * blockDim.x;
    if (i == 0) *counter = 0u;
    for (size_t k = i; k < 2u * HID; k += stride) hbuf[k] = 0.0f;   // h double buffer
    const size_t NW = (size_t)G4H * HID;                            // 4M
    for (size_t k = i; k < NW; k += stride) wbf[k] = f2bf(W_ih[k]);
    for (size_t k = i; k < NW; k += stride) {
        const size_t row = k >> 10, col = k & 1023u;
        xbf[k] = f2bf(emb[(size_t)tokens[row] * HID + col]);
    }
}

// ---------------------------------------------------------------------------
// Kernel 2: gates_x[L][4H] = x_bf16 @ W_ih_bf16^T + (b_ih + b_hh)
// One wave per 32x32 output block: 2x2 register tiling of 16x16 WMMA tiles,
// each A/B fragment reused twice -> 16 FLOP/B from cache.
// ---------------------------------------------------------------------------
__global__ void __launch_bounds__(256)
gates_gemm_wmma(const unsigned short* __restrict__ xbf,
                const unsigned short* __restrict__ wbf,
                const float* __restrict__ b_ih, const float* __restrict__ b_hh,
                float* __restrict__ gates)
{
    const int lane = threadIdx.x & 31;
    const int wave = threadIdx.x >> 5;
    const int tile = blockIdx.x * 8 + wave;      // 16384 wave-jobs (128x128 blocks)
    const int tm = tile >> 7;                    // 0..127 : 32 timesteps each
    const int tn = tile & 127;                   // 0..127 : 32 gate cols each

    // A fragment addressing (16x32 bf16, ISA layout)
    const int arow0 = tm * 32 + (lane & 15);
    const int kb    = (lane >> 4) * 8;
    // B fragment addressing (32x16 bf16): lane = N column, same K order as A
    const int bcol0 = tn * 32 + (lane & 15);
    const int khalf = (lane >> 4) * 16;

    const unsigned short* aptr0 = xbf + (size_t)arow0 * HID;
    const unsigned short* aptr1 = aptr0 + (size_t)16 * HID;
    const unsigned short* bptr0 = wbf + (size_t)bcol0 * HID;
    const unsigned short* bptr1 = bptr0 + (size_t)16 * HID;

    v8f c00 = {}, c01 = {}, c10 = {}, c11 = {};
    for (int k0 = 0; k0 < HID; k0 += 32) {
        __builtin_prefetch((const void*)(aptr0 + k0 + 256), 0, 3);  // WGP-scope
        __builtin_prefetch((const void*)(aptr1 + k0 + 256), 0, 3);
        __builtin_prefetch((const void*)(bptr0 + k0 + 256), 0, 3);
        __builtin_prefetch((const void*)(bptr1 + k0 + 256), 0, 3);
        union { uint4 u[2]; v16bf v; } A0, A1, B0, B1;
        A0.u[0] = *(const uint4*)(aptr0 + k0 + kb);        // K = kb..kb+7
        A0.u[1] = *(const uint4*)(aptr0 + k0 + 16 + kb);   // K = 16+kb..
        A1.u[0] = *(const uint4*)(aptr1 + k0 + kb);
        A1.u[1] = *(const uint4*)(aptr1 + k0 + 16 + kb);
        B0.u[0] = *(const uint4*)(bptr0 + k0 + kb);        // K = kb..kb+7
        B0.u[1] = *(const uint4*)(bptr0 + k0 + 16 + kb);   // K = 16+kb..
        B1.u[0] = *(const uint4*)(bptr1 + k0 + kb);
        B1.u[1] = *(const uint4*)(bptr1 + k0 + 16 + kb);
        c00 = __builtin_amdgcn_wmma_f32_16x16x32_bf16(false, A0.v, false, B0.v,
                                                      (short)0, c00, false, false);
        c01 = __builtin_amdgcn_wmma_f32_16x16x32_bf16(false, A0.v, false, B1.v,
                                                      (short)0, c01, false, false);
        c10 = __builtin_amdgcn_wmma_f32_16x16x32_bf16(false, A1.v, false, B0.v,
                                                      (short)0, c10, false, false);
        c11 = __builtin_amdgcn_wmma_f32_16x16x32_bf16(false, A1.v, false, B1.v,
                                                      (short)0, c11, false, false);
    }

    // C layout: lane 0-15 -> N=lane, M=j ; lane 16-31 -> N=lane-16, M=j+8
    const int coln0 = tn * 32 + (lane & 15);
    const int coln1 = coln0 + 16;
    const int row0  = tm * 32 + (lane >> 4) * 8;
    const int row1  = row0 + 16;
    const float bias0 = b_ih[coln0] + b_hh[coln0];
    const float bias1 = b_ih[coln1] + b_hh[coln1];
#pragma unroll
    for (int j = 0; j < 8; ++j) {
        gates[(size_t)(row0 + j) * G4H + coln0] = c00[j] + bias0;
        gates[(size_t)(row0 + j) * G4H + coln1] = c01[j] + bias1;
        gates[(size_t)(row1 + j) * G4H + coln0] = c10[j] + bias0;
        gates[(size_t)(row1 + j) * G4H + coln1] = c11[j] + bias1;
    }
}

// ---------------------------------------------------------------------------
// Kernel 3: persistent LSTM recurrence. 64 WGs x 256 threads.
// WG w owns hidden slice j in [16w,16w+16): W_hh rows {j, j+1024, j+2048, j+3072}
// held resident in LDS (256 KB fp32). One grid sync per timestep.
// ---------------------------------------------------------------------------
__global__ void __launch_bounds__(256)
lstm_recurrence(const float* __restrict__ W_hh, const float* __restrict__ gates,
                float* __restrict__ hbuf, float* __restrict__ hs,
                unsigned int* __restrict__ counter)
{
    const int w = blockIdx.x;        // 0..63
    const int t = threadIdx.x;       // 0..255
    extern __shared__ float smem[];
    float* Wl = smem;                // 64*1024 fp32 = 256 KB
    float* hl = smem + 64 * HID;     // 1024
    float* gb = hl + HID;            // 64 gate pre-activations
    float* cb = gb + 64;             // 16 cell states

    // Stage this WG's 64 W_hh rows (scattered gate rows) into LDS, float4 wide.
    for (int idx = t; idx < (64 * HID) / 4; idx += 256) {
        const int e = idx * 4;
        const int r = e >> 10, k = e & 1023;
        const int grow = (r >> 4) * HID + w * 16 + (r & 15);
        ((float4*)Wl)[idx] = *(const float4*)(W_hh + (size_t)grow * HID + k);
    }
    if (t < 16) cb[t] = 0.0f;
    __syncthreads();

    const int r    = t >> 2;                         // local row 0..63
    const int q    = t & 3;                          // K quarter
    const int kbse = q * 256;
    const int gq   = r >> 4;                         // gate id 0..3 (i,f,g,o)
    const int jl   = r & 15;
    const int grow = gq * HID + w * 16 + jl;         // global gate row
    const float* Wrow = Wl + r * HID + kbse;

    for (int step = 0; step < L_SEQ; ++step) {
        // stage h_prev (4 KB) into LDS
        const float* hprev = hbuf + (size_t)(step & 1) * HID;
        for (int idx = t; idx < HID / 4; idx += 256)
            ((float4*)hl)[idx] = ((const float4*)hprev)[idx];
        __syncthreads();

        // 64 dot products of length 1024, 4 threads per row
        float acc = 0.0f;
        const float4* wv = (const float4*)Wrow;
        const float4* hv = (const float4*)(hl + kbse);
#pragma unroll 8
        for (int i = 0; i < 64; ++i) {
            const float4 a = wv[i], b = hv[i];
            acc = fmaf(a.x, b.x, acc); acc = fmaf(a.y, b.y, acc);
            acc = fmaf(a.z, b.z, acc); acc = fmaf(a.w, b.w, acc);
        }
        acc += __shfl_xor(acc, 1, 32);   // combine the 4 K-quarters (adjacent lanes)
        acc += __shfl_xor(acc, 2, 32);
        if (q == 0) gb[r] = acc + gates[(size_t)step * G4H + grow];
        __syncthreads();

        if (t < 16) {   // elementwise LSTM cell for 16 hidden units
            const float gi = gb[t], gf = gb[16 + t], gg = gb[32 + t], go = gb[48 + t];
            const float iv = 1.0f / (1.0f + __expf(-gi));
            const float fv = 1.0f / (1.0f + __expf(-gf));
            const float gv = tanhf(gg);
            const float ov = 1.0f / (1.0f + __expf(-go));
            const float c  = fv * cb[t] + iv * gv;
            cb[t] = c;
            const float h = ov * tanhf(c);
            const int gj = w * 16 + t;
            hbuf[(size_t)((step + 1) & 1) * HID + gj] = h;
            hs[(size_t)step * HID + gj] = h;
        }

        // ---- grid-wide sync (monotonic counter, release/acquire) ----
        __threadfence();
        __syncthreads();
        if (t == 0) {
            __hip_atomic_fetch_add(counter, 1u, __ATOMIC_ACQ_REL,
                                   __HIP_MEMORY_SCOPE_AGENT);
            const unsigned int tgt = (unsigned int)(step + 1) * NWG;
            while (__hip_atomic_load(counter, __ATOMIC_ACQUIRE,
                                     __HIP_MEMORY_SCOPE_AGENT) < tgt)
                __builtin_amdgcn_s_sleep(2);
        }
        __syncthreads();
    }
}

// ---------------------------------------------------------------------------
// Kernel 4: tag_space = hs @ W_out^T + b_out, then row log_softmax.
// One block (64 threads) per timestep; hs row staged in LDS; W_out L2-resident.
// ---------------------------------------------------------------------------
__global__ void __launch_bounds__(64)
tag_logsoftmax(const float* __restrict__ hs, const float* __restrict__ W_out,
               const float* __restrict__ b_out, float* __restrict__ out)
{
    const int t = blockIdx.x;
    const int j = threadIdx.x;
    __shared__ float hrow[HID];
    __shared__ float vals[NTAG];
    __shared__ float red[2];

    for (int idx = j; idx < HID / 4; idx += 64)
        ((float4*)hrow)[idx] = ((const float4*)(hs + (size_t)t * HID))[idx];
    __syncthreads();

    if (j < NTAG) {
        float acc = b_out[j];
        const float* wr = W_out + (size_t)j * HID;
        for (int k = 0; k < HID; k += 4) {
            acc = fmaf(hrow[k + 0], wr[k + 0], acc);
            acc = fmaf(hrow[k + 1], wr[k + 1], acc);
            acc = fmaf(hrow[k + 2], wr[k + 2], acc);
            acc = fmaf(hrow[k + 3], wr[k + 3], acc);
        }
        vals[j] = acc;
    }
    __syncthreads();
    if (j == 0) {
        float m = vals[0];
        for (int i = 1; i < NTAG; ++i) m = fmaxf(m, vals[i]);
        float s = 0.0f;
        for (int i = 0; i < NTAG; ++i) s += __expf(vals[i] - m);
        red[0] = m;
        red[1] = logf(s);
    }
    __syncthreads();
    if (j < NTAG) out[(size_t)t * NTAG + j] = vals[j] - red[0] - red[1];
}

// ---------------------------------------------------------------------------
extern "C" void kernel_launch(void* const* d_in, const int* in_sizes, int n_in,
                              void* d_out, int out_size, void* d_ws, size_t ws_size,
                              hipStream_t stream)
{
    const int*   tokens = (const int*)  d_in[0];
    const float* emb    = (const float*)d_in[1];
    const float* W_ih   = (const float*)d_in[2];
    const float* W_hh   = (const float*)d_in[3];
    const float* b_ih   = (const float*)d_in[4];
    const float* b_hh   = (const float*)d_in[5];
    const float* W_out  = (const float*)d_in[6];
    const float* b_out  = (const float*)d_in[7];
    float* out = (float*)d_out;

    char* ws = (char*)d_ws;
    unsigned short* xbf   = (unsigned short*)(ws);                    //  8 MB
    unsigned short* wbf   = (unsigned short*)(ws + (8u  << 20));      //  8 MB
    float*          gates = (float*)        (ws + (16u << 20));       // 64 MB
    float*          hs    = (float*)        (ws + (80u << 20));       // 16 MB
    float*          hbuf  = (float*)        (ws + (96u << 20));       //  8 KB
    unsigned int*   cnt   = (unsigned int*) (ws + (96u << 20) + 8192);

    prep_kernel<<<2048, 256, 0, stream>>>(tokens, emb, W_ih, xbf, wbf, hbuf, cnt);

    // 128x128 blocks of 32x32 wave-tiles, 8 waves per 256-thread block
    gates_gemm_wmma<<<(128 * 128) / 8, 256, 0, stream>>>(xbf, wbf, b_ih, b_hh, gates);

    const size_t smem = (size_t)(64 * HID + HID + 64 + 16) * sizeof(float); // ~260 KB
    lstm_recurrence<<<NWG, 256, smem, stream>>>(W_hh, gates, hbuf, hs, cnt);

    tag_logsoftmax<<<L_SEQ, 64, 0, stream>>>(hs, W_out, b_out, out);
}
